// AttentionBlock_19851338842651
// MI455X (gfx1250) — matrix operand order repaired and checked by reading, not compiled, not run on hardware
//
#include <hip/hip_runtime.h>
#include <hip/hip_bf16.h>

typedef __attribute__((ext_vector_type(16))) _Float16 v16h;
typedef __attribute__((ext_vector_type(8)))  float    v8f;
typedef unsigned int uint;

#define Bb   4
#define Cc   128
#define HWn  4096
#define NTOK (Bb * HWn)   // 16384 tokens

union AFrag {
    v16h v;
    uint u[8];
    _Float16 h[16];
};

__device__ __forceinline__ v8f wmma_f16(v16h a, v16h b, v8f c) {
    // D = A(16x32 f16) * B(32x16 f16) + C(16x16 f32)
    return __builtin_amdgcn_wmma_f32_16x16x32_f16(
        false, a, false, b, (short)0, c, false, false);
}

// A fragment: 16x32 tile from row-major f16 src (leading dim ld).
// ISA layout: lane L -> row M = L%16, half = L/16; VGPR v holds K pair at
// K = (v&4)*4 + half*8 + (v&3)*2.
__device__ __forceinline__ v16h load_a16x32_rowmajor(const _Float16* src, int ld,
                                                     int rowBase, int kBase, int lane) {
    const int m = lane & 15, half = lane >> 4;
    AFrag f;
    const _Float16* p = src + (size_t)(rowBase + m) * ld + kBase;
#pragma unroll
    for (int v = 0; v < 8; ++v) {
        const int k = ((v & 4) << 2) + half * 8 + ((v & 3) << 1);
        f.u[v] = *reinterpret_cast<const uint*>(p + k);   // even K -> 4B aligned
    }
    return f.v;
}

// B fragment (32x16, K x N) where B[k][n] = src[n*ld + k]  (n-major source, i.e. B = src^T).
// ISA layout: lane -> col N = lane%16, half = lane/16; VGPR v holds K pair at
// K = (v&4)*4 + half*8 + (v&3)*2.
__device__ __forceinline__ v16h load_b32x16_from_nmajor(const _Float16* src, int ld,
                                                        int nBase, int kBase, int lane) {
    const int n = lane & 15, kh = (lane >> 4) << 3;
    AFrag f;
    const _Float16* p = src + (size_t)(nBase + n) * ld + kBase + kh;
#pragma unroll
    for (int v = 0; v < 8; ++v)
        f.u[v] = *reinterpret_cast<const uint*>(p + ((v & 4) << 2) + ((v & 3) << 1));
    return f.v;
}

// Async copy of 16 bytes per lane: global -> LDS, tracked by ASYNCcnt.
__device__ __forceinline__ void async_copy16(const void* gptr, void* lptr) {
    const uint lds = (uint)(uintptr_t)lptr;              // low 32 bits = LDS byte addr
    const unsigned long long ga = (unsigned long long)(uintptr_t)gptr;
    asm volatile("global_load_async_to_lds_b128 %0, %1, off"
                 :: "v"(lds), "v"(ga) : "memory");
}

__device__ __forceinline__ void wait_async_all() {
    asm volatile("s_wait_asynccnt 0x0" ::: "memory");
}

// ---------------------------------------------------------------------------
// 1) GroupNorm statistics: one block per (batch, group). 16 channels * 4096 px.
// ---------------------------------------------------------------------------
__global__ void __launch_bounds__(256) gn_stats_kernel(const float* __restrict__ x,
                                                       float* __restrict__ stats) {
    const int bg = blockIdx.x;            // 0..31
    const int tid = threadIdx.x;
    const float* base = x + (size_t)bg * 16 * HWn;  // 16 contiguous channels
    float s = 0.f, s2 = 0.f;
    for (int i = tid; i < 16 * HWn; i += 256) {
        float v = base[i];
        s += v; s2 += v * v;
    }
    __shared__ float r1[256], r2[256];
    r1[tid] = s; r2[tid] = s2;
    __syncthreads();
    for (int off = 128; off > 0; off >>= 1) {
        if (tid < off) { r1[tid] += r1[tid + off]; r2[tid] += r2[tid + off]; }
        __syncthreads();
    }
    if (tid == 0) {
        const float n = 16.f * HWn;
        const float mean = r1[0] / n;
        const float var  = r2[0] / n - mean * mean;
        stats[bg * 2]     = mean;
        stats[bg * 2 + 1] = rsqrtf(var + 1e-5f);
    }
}

// ---------------------------------------------------------------------------
// 2) Apply GroupNorm, cast to f16, transpose [B,C,HW] -> token-major [B*HW, C]
// ---------------------------------------------------------------------------
__global__ void __launch_bounds__(256) gn_apply_kernel(const float* __restrict__ x,
                                                       const float* __restrict__ nw,
                                                       const float* __restrict__ nb,
                                                       const float* __restrict__ stats,
                                                       _Float16* __restrict__ h16) {
    const int idx = blockIdx.x * 256 + threadIdx.x;   // over B*C*HW = 2^21
    if (idx >= Bb * Cc * HWn) return;
    const int hw = idx & (HWn - 1);
    const int c  = (idx >> 12) & (Cc - 1);
    const int b  = idx >> 19;
    const int g  = c >> 4;
    const float mean = stats[(b * 8 + g) * 2];
    const float rstd = stats[(b * 8 + g) * 2 + 1];
    const float v = (x[idx] - mean) * rstd * nw[c] + nb[c];
    h16[(size_t)(b * HWn + hw) * Cc + c] = (_Float16)v;
}

// ---------------------------------------------------------------------------
// 3) Cast weights to f16 (qkv_w: 384x128, proj_w: 128x128, both row-major)
// ---------------------------------------------------------------------------
__global__ void __launch_bounds__(256) cvt_w_kernel(const float* __restrict__ qkvw,
                                                    const float* __restrict__ projw,
                                                    _Float16* __restrict__ qw16,
                                                    _Float16* __restrict__ pw16) {
    const int idx = blockIdx.x * 256 + threadIdx.x;
    if (idx < 3 * Cc * Cc) qw16[idx] = (_Float16)qkvw[idx];
    if (idx < Cc * Cc)     pw16[idx] = (_Float16)projw[idx];
}

// ---------------------------------------------------------------------------
// 4) QKV GEMM: [16384 x 128] x [128 x 384] -> f16 Q, K (token-major) and
//    V transposed (channel-major, [B][C][HW]) so the attention kernel's
//    P @ V B-fragments are contiguous-per-lane LDS loads.
// ---------------------------------------------------------------------------
__global__ void __launch_bounds__(256) qkv_gemm_kernel(const _Float16* __restrict__ h16,
                                                       const _Float16* __restrict__ w16,
                                                       const float* __restrict__ bias,
                                                       _Float16* __restrict__ q16,
                                                       _Float16* __restrict__ k16,
                                                       _Float16* __restrict__ vT16) {
    const int lane = threadIdx.x & 31;
    const int wave = threadIdx.x >> 5;
    const int gw = blockIdx.x * 8 + wave;     // 24576 wave-tiles
    const int tn = gw % 24;                   // N tiles: 384/16
    const int tm = gw / 24;                   // M tiles: 16384/16
    v8f acc = {};
#pragma unroll
    for (int kc = 0; kc < 4; ++kc) {
        v16h a  = load_a16x32_rowmajor(h16, Cc, tm * 16, kc * 32, lane);
        v16h bm = load_b32x16_from_nmajor(w16, Cc, tn * 16, kc * 32, lane);  // B = W^T
        acc = wmma_f16(a, bm, acc);
    }
    const int col  = tn * 16 + (lane & 15);
    const int half = lane >> 4;
    const float bv = bias[col];
#pragma unroll
    for (int r = 0; r < 8; ++r) {
        const int tok = tm * 16 + r + half * 8;   // C/D layout: VGPR r -> row r + half*8
        const _Float16 val = (_Float16)(acc[r] + bv);
        if (col < Cc) {
            q16[(size_t)tok * Cc + col] = val;
        } else if (col < 2 * Cc) {
            k16[(size_t)tok * Cc + (col - Cc)] = val;
        } else {
            const int b = tok >> 12, thw = tok & (HWn - 1);
            vT16[(size_t)(b * Cc + (col - 2 * Cc)) * HWn + thw] = val;
        }
    }
}

// ---------------------------------------------------------------------------
// 5) Flash attention. 8 waves / block; wave owns 16 queries; block = 128
//    queries. 32-key K/V chunks staged into LDS via async-to-LDS DMA
//    (ASYNCcnt). Online softmax on the fp32 accumulators; P goes through a
//    per-wave LDS tile to get the A-fragment layout for P @ V.
// ---------------------------------------------------------------------------
__global__ void __launch_bounds__(256) flash_attn_kernel(const _Float16* __restrict__ q16,
                                                         const _Float16* __restrict__ k16,
                                                         const _Float16* __restrict__ vT16,
                                                         _Float16* __restrict__ o16) {
    const int lane = threadIdx.x & 31;
    const int wave = threadIdx.x >> 5;
    const int m16 = lane & 15, half = lane >> 4;
    const int b    = blockIdx.x >> 5;          // 4 batches * 32 q-blocks
    const int qblk = blockIdx.x & 31;
    const int qbase = qblk * 128 + wave * 16;
    const size_t tokBase = (size_t)b * HWn;

    __shared__ alignas(16) _Float16 Ks[32 * Cc];       // 8 KB  [key][chan]
    __shared__ alignas(16) _Float16 VsT[Cc * 32];      // 8 KB  [chan][key]
    __shared__ alignas(16) _Float16 Ps[8][16 * 32];    // 8 KB  per-wave P tiles

    // Q fragments held in registers for the whole pass (16 x 128 f16 = 32 VGPRs)
    v16h qf[4];
#pragma unroll
    for (int kc = 0; kc < 4; ++kc)
        qf[kc] = load_a16x32_rowmajor(q16 + tokBase * Cc, Cc, qbase, kc * 32, lane);

    v8f o[8] = {};                 // 16 x 128 fp32 output accumulator
    float mrow[8], lrow[8];
#pragma unroll
    for (int r = 0; r < 8; ++r) { mrow[r] = -3.0e38f; lrow[r] = 0.f; }

    const float scale = 0.08838834764831845f;  // 1/sqrt(128)
    const _Float16* gvb = vT16 + (size_t)b * Cc * HWn;  // batch base, channel-major

    for (int kb = 0; kb < HWn; kb += 32) {
        // --- async DMA the 32-key K (token-major) and V^T (channel-major)
        //     chunks into LDS: 512 x 16B each, bypassing VGPRs ---
        {
            const uint4* gk = reinterpret_cast<const uint4*>(k16 + (tokBase + kb) * Cc);
            uint4* sk = reinterpret_cast<uint4*>(Ks);
            uint4* sv = reinterpret_cast<uint4*>(VsT);
            for (int i = threadIdx.x; i < 512; i += 256) {
                async_copy16(gk + i, sk + i);
                const int row = i >> 2, seg = i & 3;   // 128 chans x 4 x 16B
                const uint4* g = reinterpret_cast<const uint4*>(gvb + (size_t)row * HWn + kb) + seg;
                async_copy16(g, sv + i);
            }
            if (kb + 32 < HWn) {   // prefetch next K chunk into cache
                __builtin_prefetch(gk + 512, 0, 1);
            }
        }
        wait_async_all();
        __syncthreads();

        // S = Q * K^T for 32 keys: two 16x16 fp32 accumulators
        v8f s0 = {}, s1 = {};
#pragma unroll
        for (int kc = 0; kc < 4; ++kc) {
            v16h b0 = load_b32x16_from_nmajor(Ks, Cc, 0,  kc * 32, lane);
            v16h b1 = load_b32x16_from_nmajor(Ks, Cc, 16, kc * 32, lane);
            s0 = wmma_f16(qf[kc], b0, s0);
            s1 = wmma_f16(qf[kc], b1, s1);
        }

        // online softmax across these 32 keys
        float alpha[8];
#pragma unroll
        for (int r = 0; r < 8; ++r) {
            float v0 = s0[r] * scale, v1 = s1[r] * scale;
            float mx = fmaxf(v0, v1);
#pragma unroll
            for (int d = 1; d < 16; d <<= 1) mx = fmaxf(mx, __shfl_xor(mx, d, 32));
            const float mnew = fmaxf(mrow[r], mx);
            const float a  = __expf(mrow[r] - mnew);
            const float p0 = __expf(v0 - mnew);
            const float p1 = __expf(v1 - mnew);
            float rs = p0 + p1;
#pragma unroll
            for (int d = 1; d < 16; d <<= 1) rs += __shfl_xor(rs, d, 32);
            lrow[r] = lrow[r] * a + rs;
            mrow[r] = mnew;
            alpha[r] = a;
            s0[r] = p0; s1[r] = p1;
        }
#pragma unroll
        for (int n = 0; n < 8; ++n)
#pragma unroll
            for (int r = 0; r < 8; ++r) o[n][r] *= alpha[r];

        // P (16x32) -> per-wave LDS tile, reload in A-fragment layout
        _Float16* pb = &Ps[wave][0];
#pragma unroll
        for (int r = 0; r < 8; ++r) {
            const int row = r + half * 8;
            pb[row * 32 + m16]      = (_Float16)s0[r];
            pb[row * 32 + 16 + m16] = (_Float16)s1[r];
        }
        __builtin_amdgcn_wave_barrier();
        asm volatile("" ::: "memory");     // LDS ops stay in order within a wave
        v16h pf = load_a16x32_rowmajor(pb, 32, 0, 0, lane);

        // O += P * V  (8 channel subtiles; V^T is n-major -> contiguous lane loads)
#pragma unroll
        for (int n = 0; n < 8; ++n) {
            v16h vf = load_b32x16_from_nmajor(VsT, 32, n * 16, 0, lane);
            o[n] = wmma_f16(pf, vf, o[n]);
        }
        __syncthreads();   // protect LDS K/V before next chunk overwrites
    }

    // epilogue: divide by l, store token-major f16
#pragma unroll
    for (int r = 0; r < 8; ++r) {
        const float inv = 1.f / lrow[r];
        const int tok = qbase + r + half * 8;
        _Float16* dst = o16 + (tokBase + tok) * Cc + m16;
#pragma unroll
        for (int n = 0; n < 8; ++n)
            dst[n * 16] = (_Float16)(o[n][r] * inv);
    }
}

// ---------------------------------------------------------------------------
// 6) Proj GEMM + bias + residual, output fp32 in [B,C,H,W] layout.
// ---------------------------------------------------------------------------
__global__ void __launch_bounds__(256) proj_gemm_kernel(const _Float16* __restrict__ o16,
                                                        const _Float16* __restrict__ pw16,
                                                        const float* __restrict__ pbias,
                                                        const float* __restrict__ x,
                                                        float* __restrict__ out) {
    const int lane = threadIdx.x & 31;
    const int wave = threadIdx.x >> 5;
    const int gw = blockIdx.x * 8 + wave;   // 8192 wave-tiles
    const int tn = gw & 7;                  // N tiles: 128/16
    const int tm = gw >> 3;                 // M tiles: 16384/16
    v8f acc = {};
#pragma unroll
    for (int kc = 0; kc < 4; ++kc) {
        v16h a  = load_a16x32_rowmajor(o16, Cc, tm * 16, kc * 32, lane);
        v16h bm = load_b32x16_from_nmajor(pw16, Cc, tn * 16, kc * 32, lane);  // B = W^T
        acc = wmma_f16(a, bm, acc);
    }
    const int col  = tn * 16 + (lane & 15);
    const int half = lane >> 4;
    const float bv = pbias[col];
#pragma unroll
    for (int r = 0; r < 8; ++r) {
        const int tok = tm * 16 + r + half * 8;
        const int bb  = tok >> 12;             // token / HW
        const int thw = tok & (HWn - 1);
        const size_t oi = (size_t)(bb * Cc + col) * HWn + thw;
        out[oi] = x[oi] + acc[r] + bv;
    }
}

// ---------------------------------------------------------------------------
extern "C" void kernel_launch(void* const* d_in, const int* in_sizes, int n_in,
                              void* d_out, int out_size, void* d_ws, size_t ws_size,
                              hipStream_t stream) {
    const float* x      = (const float*)d_in[0];
    const float* norm_w = (const float*)d_in[1];
    const float* norm_b = (const float*)d_in[2];
    const float* qkv_w  = (const float*)d_in[3];
    const float* qkv_b  = (const float*)d_in[4];
    const float* proj_w = (const float*)d_in[5];
    const float* proj_b = (const float*)d_in[6];
    float* out = (float*)d_out;

    // workspace layout (all 256B aligned)
    char* w = (char*)d_ws;
    float*    stats = (float*)w;                               // 256 B
    _Float16* h16   = (_Float16*)(w + 256);                    // 4 MB
    _Float16* q16   = h16 + (size_t)NTOK * Cc;                 // 4 MB
    _Float16* k16   = q16 + (size_t)NTOK * Cc;                 // 4 MB
    _Float16* vT16  = k16 + (size_t)NTOK * Cc;                 // 4 MB (channel-major)
    _Float16* o16   = vT16 + (size_t)NTOK * Cc;                // 4 MB
    _Float16* qw16  = o16 + (size_t)NTOK * Cc;                 // 96 KB
    _Float16* pw16  = qw16 + 3 * Cc * Cc;                      // 32 KB

    gn_stats_kernel<<<32, 256, 0, stream>>>(x, stats);
    gn_apply_kernel<<<(Bb * Cc * HWn) / 256, 256, 0, stream>>>(x, norm_w, norm_b, stats, h16);
    cvt_w_kernel<<<(3 * Cc * Cc + 255) / 256, 256, 0, stream>>>(qkv_w, proj_w, qw16, pw16);
    qkv_gemm_kernel<<<(NTOK / 16) * 24 / 8, 256, 0, stream>>>(h16, qw16, qkv_b, q16, k16, vT16);
    flash_attn_kernel<<<Bb * (HWn / 128), 256, 0, stream>>>(q16, k16, vT16, o16);
    proj_gemm_kernel<<<(NTOK / 16) * 8 / 8, 256, 0, stream>>>(o16, pw16, proj_b, x, out);
}
